// RoiAlign_32366873543210
// MI455X (gfx1250) — matrix operand order repaired and checked by reading, not compiled, not run on hardware
//
#include <hip/hip_runtime.h>

typedef _Float16 h4   __attribute__((ext_vector_type(4)));
typedef _Float16 v8h  __attribute__((ext_vector_type(8)));
typedef _Float16 v16h __attribute__((ext_vector_type(16)));
typedef float    v8f  __attribute__((ext_vector_type(8)));

#define IMG_H 128
#define IMG_W 128
#define NCH   256
#define OUTN  32
#define TSTR  136   // padded row stride in f16 (272B) to de-align LDS banks

__global__ __launch_bounds__(256)
void roi_align_wmma(const float* __restrict__ img,
                    const float* __restrict__ bbox,
                    float* __restrict__ out)
{
    __shared__ int   s_x0[OUTN];
    __shared__ float s_wx0[OUTN], s_wx1[OUTN];
    __shared__ int   s_y0[OUTN];
    __shared__ float s_wy0[OUTN], s_wy1[OUTN];
    __shared__ __align__(16) _Float16 s_Wx[OUTN][TSTR];        // dense x-interp matrix
    __shared__ __align__(16) _Float16 s_tmp[8][OUTN][TSTR];    // per-wave y-combined rows

    const int b    = blockIdx.x >> 5;       // 64 batches
    const int cg   = blockIdx.x & 31;       // 32 channel-groups of 8
    const int tid  = threadIdx.x;
    const int wave = tid >> 5;
    const int lane = tid & 31;

    // ---- zero Wx (4352 halves, 17 per thread) ----
    {
        _Float16* p = &s_Wx[0][0];
        for (int k = tid; k < OUTN * TSTR; k += 256) p[k] = (_Float16)0.0f;
    }

    // ---- per-batch separable bilinear weights (x for i, y for j) ----
    if (tid < 64) {
        const int   idx = tid & 31;
        const float x1 = bbox[b * 4 + 0], y1 = bbox[b * 4 + 1];
        const float x2 = bbox[b * 4 + 2], y2 = bbox[b * 4 + 3];
        if (tid < 32) {
            float g  = ((float)idx * (x2 - x1) + x1 - 0.5f) * 2.0f;   // gx(i)
            float ix = (g + 1.0f) * (0.5f * (float)IMG_W) - 0.5f;
            float f  = floorf(ix);
            s_x0[idx]  = (int)f;
            s_wx1[idx] = ix - f;
            s_wx0[idx] = 1.0f - (ix - f);
        } else {
            float g  = ((float)idx * (y2 - y1) + y1 - 0.5f) * 2.0f;   // gy(j)
            float iy = (g + 1.0f) * (0.5f * (float)IMG_H) - 0.5f;
            float f  = floorf(iy);
            s_y0[idx]  = (int)f;
            s_wy1[idx] = iy - f;
            s_wy0[idx] = 1.0f - (iy - f);
        }
    }
    __syncthreads();

    // ---- scatter the two nonzeros per row into dense Wx (zero-padding OOB) ----
    if (tid < 32) {
        const int x0 = s_x0[tid];
        if (x0 >= 0     && x0     < IMG_W) s_Wx[tid][x0]     = (_Float16)s_wx0[tid];
        if (x0 + 1 >= 0 && x0 + 1 < IMG_W) s_Wx[tid][x0 + 1] = (_Float16)s_wx1[tid];
    }

    // ---- stage 1: y-contraction (2-term, sparse) with fully coalesced row loads ----
    const int    c    = cg * 8 + wave;
    const float* imgc = img + ((size_t)(b * NCH + c)) * (IMG_H * IMG_W);
    const int    wcol = lane * 4;           // 32 lanes x float4 = full 128-wide row

    for (int j = 0; j < OUTN; ++j) {
        const int   y0 = s_y0[j];
        const float m0 = (y0 >= 0     && y0     < IMG_H) ? s_wy0[j] : 0.0f;
        const float m1 = (y0 + 1 >= 0 && y0 + 1 < IMG_H) ? s_wy1[j] : 0.0f;
        const int   y0c = (y0 < 0) ? 0 : ((y0 > IMG_H - 1) ? IMG_H - 1 : y0);
        const int   y1c = (y0 + 1 < 0) ? 0 : ((y0 + 1 > IMG_H - 1) ? IMG_H - 1 : y0 + 1);

        const float4 r0 = *(const float4*)(imgc + (size_t)y0c * IMG_W + wcol);
        const float4 r1 = *(const float4*)(imgc + (size_t)y1c * IMG_W + wcol);

        h4 hv;
        hv[0] = (_Float16)(m0 * r0.x + m1 * r1.x);
        hv[1] = (_Float16)(m0 * r0.y + m1 * r1.y);
        hv[2] = (_Float16)(m0 * r0.z + m1 * r1.z);
        hv[3] = (_Float16)(m0 * r0.w + m1 * r1.w);
        *(h4*)&s_tmp[wave][j][wcol] = hv;
    }
    __syncthreads();   // Wx visible + tmp committed before fragment loads

    // ---- stage 2: Out(32x32) = Wx(32x128) x Tmp^T(128x32) via v_wmma_f32_16x16x32_f16 ----
    v8f acc00 = {}, acc01 = {}, acc10 = {}, acc11 = {};
    const int half = lane >> 4;     // 0: lanes 0-15, 1: lanes 16-31
    const int lrow = lane & 15;

    #pragma unroll
    for (int kk = 0; kk < IMG_W; kk += 32) {
        // A fragments (16-bit A layout: lane<16 -> K {0..7,16..23}, lane>=16 -> +8)
        const int ka = kk + half * 8;
        const v8h a0lo = *(const v8h*)&s_Wx[lrow][ka];
        const v8h a0hi = *(const v8h*)&s_Wx[lrow][ka + 16];
        const v8h a1lo = *(const v8h*)&s_Wx[16 + lrow][ka];
        const v8h a1hi = *(const v8h*)&s_Wx[16 + lrow][ka + 16];
        const v16h A0 = __builtin_shufflevector(a0lo, a0hi, 0,1,2,3,4,5,6,7,8,9,10,11,12,13,14,15);
        const v16h A1 = __builtin_shufflevector(a1lo, a1hi, 0,1,2,3,4,5,6,7,8,9,10,11,12,13,14,15);

        // B fragments (B layout: N = lane&15, K {0..7,16..23} for lane<16, +8 for lane>=16)
        const int kb = kk + half * 8;
        const v8h b0lo = *(const v8h*)&s_tmp[wave][lrow][kb];
        const v8h b0hi = *(const v8h*)&s_tmp[wave][lrow][kb + 16];
        const v8h b1lo = *(const v8h*)&s_tmp[wave][16 + lrow][kb];
        const v8h b1hi = *(const v8h*)&s_tmp[wave][16 + lrow][kb + 16];
        const v16h B0 = __builtin_shufflevector(b0lo, b0hi, 0,1,2,3,4,5,6,7,8,9,10,11,12,13,14,15);
        const v16h B1 = __builtin_shufflevector(b1lo, b1hi, 0,1,2,3,4,5,6,7,8,9,10,11,12,13,14,15);

        acc00 = __builtin_amdgcn_wmma_f32_16x16x32_f16(false, A0, false, B0, (short)0, acc00, false, false);
        acc01 = __builtin_amdgcn_wmma_f32_16x16x32_f16(false, A0, false, B1, (short)0, acc01, false, false);
        acc10 = __builtin_amdgcn_wmma_f32_16x16x32_f16(false, A1, false, B0, (short)0, acc10, false, false);
        acc11 = __builtin_amdgcn_wmma_f32_16x16x32_f16(false, A1, false, B1, (short)0, acc11, false, false);
    }

    // ---- quantize (round-half-even like jnp.round) + coalesced stores (j across lanes) ----
    float* outc = out + ((size_t)(b * NCH + c)) * (OUTN * OUTN);
    const float q  = 255.0f / 4.0f;       // 63.75
    const float iq = 4.0f / 255.0f;
    #pragma unroll
    for (int v = 0; v < 8; ++v) {
        const int i0 = v + 8 * half;      // C layout: M = v (+8 for upper half-wave)
        const int jj = lrow;              // N = lane & 15
        outc[(i0)      * OUTN + jj]      = __builtin_rintf(acc00[v] * q) * iq;
        outc[(i0)      * OUTN + 16 + jj] = __builtin_rintf(acc01[v] * q) * iq;
        outc[(i0 + 16) * OUTN + jj]      = __builtin_rintf(acc10[v] * q) * iq;
        outc[(i0 + 16) * OUTN + 16 + jj] = __builtin_rintf(acc11[v] * q) * iq;
    }
}

extern "C" void kernel_launch(void* const* d_in, const int* in_sizes, int n_in,
                              void* d_out, int out_size, void* d_ws, size_t ws_size,
                              hipStream_t stream) {
    const float* img  = (const float*)d_in[0];   // [64,256,128,128] f32
    const float* bbox = (const float*)d_in[1];   // [64,4] f32
    float*       out  = (float*)d_out;           // [64,256,32,32] f32

    dim3 grid(64 * 32);   // batch x channel-group(8)
    dim3 block(256);      // 8 waves (wave32), one channel per wave
    hipLaunchKernelGGL(roi_align_wmma, grid, block, 0, stream, img, bbox, out);
}
